// window_attention_66408784331143
// MI455X (gfx1250) — matrix operand order repaired and checked by reading, not compiled, not run on hardware
//
#include <hip/hip_runtime.h>
#include <hip/hip_bf16.h>

typedef __attribute__((ext_vector_type(16))) __bf16 v16bf;
typedef __attribute__((ext_vector_type(8)))  __bf16 v8bf;
typedef __attribute__((ext_vector_type(8)))  float  v8f;
typedef __attribute__((ext_vector_type(4)))  float  v4f;

#define NUM_B   2048
#define SEQ_N   49
#define DIM_C   384
#define HEADS   12
#define DH      32
#define NPAD    64
#define TOKENS  (NUM_B * SEQ_N)      // 100352
#define QKV_N   (3 * DIM_C)          // 1152
#define BH      (NUM_B * HEADS)      // 24576

union Frag16 { v16bf v; v8bf h2[2]; };

// ---------------------------------------------------------------- weights cvt
__global__ void __launch_bounds__(256)
convert_w_kernel(const float* __restrict__ qkv_w, const float* __restrict__ proj_w,
                 __bf16* __restrict__ wqkv, __bf16* __restrict__ wproj) {
    int i = blockIdx.x * 256 + threadIdx.x;
    if (i < QKV_N * DIM_C) wqkv[i] = (__bf16)qkv_w[i];
    if (i < DIM_C * DIM_C) wproj[i] = (__bf16)proj_w[i];
}

// ------------------------------------------------------------- zero padding
// Zero Q/K rows 49..63 and Vt cols 49..63 so padded keys never feed NaN/garbage
// into the WMMA pipes (0 x finite = 0 after softmax).
__global__ void __launch_bounds__(256)
pad_zero_kernel(__bf16* __restrict__ Qp, __bf16* __restrict__ Kp, __bf16* __restrict__ Vt) {
    size_t base = (size_t)blockIdx.x * (NPAD * DH);   // one (b,h) per block
    for (int idx = threadIdx.x; idx < 15 * DH; idx += 256) {
        int r = SEQ_N + idx / DH, d = idx % DH;
        Qp[base + (size_t)r * DH + d] = (__bf16)0.f;
        Kp[base + (size_t)r * DH + d] = (__bf16)0.f;
        int dd = idx / 15, c = SEQ_N + idx % 15;
        Vt[base + (size_t)dd * NPAD + c] = (__bf16)0.f;
    }
}

// ------------------------------------------------------------------ QKV GEMM
// C[t, j] = x[t,:] . qkv_w[j,:] + qkv_b[j];  scatter into Qp/Kp (scaled q) / Vt.
// Block tile 64 tokens x 64 cols, 8 waves (4x2), each wave one 16x32 output tile.
__global__ void __launch_bounds__(256)
qkv_gemm_kernel(const float* __restrict__ x, const __bf16* __restrict__ wqkv,
                const float* __restrict__ qkv_b,
                __bf16* __restrict__ Qp, __bf16* __restrict__ Kp, __bf16* __restrict__ Vt) {
    __shared__ __bf16 As[64 * 32];
    __shared__ __bf16 Bs[64 * 32];
    const int mBase = blockIdx.x * 64;
    const int cBase = blockIdx.y * 64;
    const int tid   = threadIdx.x;
    const int wave  = tid >> 5, lane = tid & 31;
    const int wm = wave >> 1, wc = wave & 1;
    const int l16 = lane & 15, hi = lane >> 4;

    v8f acc0 = {}; v8f acc1 = {};

    for (int k0 = 0; k0 < DIM_C; k0 += 32) {
        // stage A (fp32 -> bf16) and B tiles into LDS: 8 elems / thread each
        {
            int idx = tid * 8;
            int r = idx >> 5, c = idx & 31;
            const float* src = x + (size_t)(mBase + r) * DIM_C + k0 + c;
            v4f f0 = *(const v4f*)src;
            v4f f1 = *(const v4f*)(src + 4);
            v8bf vv;
            vv[0]=(__bf16)f0[0]; vv[1]=(__bf16)f0[1]; vv[2]=(__bf16)f0[2]; vv[3]=(__bf16)f0[3];
            vv[4]=(__bf16)f1[0]; vv[5]=(__bf16)f1[1]; vv[6]=(__bf16)f1[2]; vv[7]=(__bf16)f1[3];
            *(v8bf*)&As[idx] = vv;
            const __bf16* bsrc = wqkv + (size_t)(cBase + r) * DIM_C + k0 + c;
            *(v8bf*)&Bs[idx] = *(const v8bf*)bsrc;
            if (k0 + 32 < DIM_C) {            // global_prefetch_b8 of next K tile
                __builtin_prefetch(src + 32, 0, 1);
                __builtin_prefetch(bsrc + 32, 0, 1);
            }
        }
        __syncthreads();
        // A fragment: K blocks of 16 per lane-half -> 16 contiguous halfs
        Frag16 af, bf0, bf1;
        {
            int row = wm * 16 + l16;
            int ko  = hi * 16;
            af.h2[0] = *(const v8bf*)&As[row * 32 + ko];
            af.h2[1] = *(const v8bf*)&As[row * 32 + ko + 8];
            // B fragment: K blocks of 16 per lane-half -> 16 contiguous halfs
            int col0 = wc * 32 + l16;
            bf0.h2[0] = *(const v8bf*)&Bs[col0 * 32 + hi * 16];
            bf0.h2[1] = *(const v8bf*)&Bs[col0 * 32 + hi * 16 + 8];
            int col1 = col0 + 16;
            bf1.h2[0] = *(const v8bf*)&Bs[col1 * 32 + hi * 16];
            bf1.h2[1] = *(const v8bf*)&Bs[col1 * 32 + hi * 16 + 8];
        }
        acc0 = __builtin_amdgcn_wmma_f32_16x16x32_bf16(false, af.v, false, bf0.v,
                                                       (short)0, acc0, false, false);
        acc1 = __builtin_amdgcn_wmma_f32_16x16x32_bf16(false, af.v, false, bf1.v,
                                                       (short)0, acc1, false, false);
        __syncthreads();
    }

    const float scale = 0.17677669529663687f;  // 32^-0.5
    #pragma unroll
    for (int sub = 0; sub < 2; ++sub) {
        const v8f& acc = sub ? acc1 : acc0;
        int j = cBase + wc * 32 + sub * 16 + l16;
        float bias = qkv_b[j];
        int sect = j / DIM_C;                   // 0=q 1=k 2=v (uniform per fragment)
        int jj = j % DIM_C;
        int h = jj >> 5, d = jj & 31;
        #pragma unroll
        for (int r = 0; r < 8; ++r) {
            int row = wm * 16 + (hi ? r + 8 : r);
            int t = mBase + row;
            int b = t / SEQ_N, n = t % SEQ_N;
            float v = acc[r] + bias;
            size_t bh = (size_t)b * HEADS + h;
            if (sect == 0)      Qp[bh * (NPAD * DH) + (size_t)n * DH + d] = (__bf16)(v * scale);
            else if (sect == 1) Kp[bh * (NPAD * DH) + (size_t)n * DH + d] = (__bf16)v;
            else                Vt[bh * (NPAD * DH) + (size_t)d * NPAD + n] = (__bf16)v;
        }
    }
}

// ----------------------------------------------------------------- attention
// One (b,h) per block; 4 waves, each owns 16 query rows. Scores via 4 WMMAs
// (Dh=32 = one K-step), masked softmax with __shfl_xor row reductions (wave32
// halves), probabilities staged through LDS to re-layout C-format -> A-format,
// then A.V with 2 K-steps x 2 d-tiles = 4 WMMAs.
__global__ void __launch_bounds__(128)
attn_kernel(const __bf16* __restrict__ Qp, const __bf16* __restrict__ Kp,
            const __bf16* __restrict__ Vt, const float* __restrict__ mask,
            __bf16* __restrict__ Ao) {
    __shared__ __bf16 Ps[NPAD * NPAD];          // 64 rows x 64 key-cols, bf16
    const int bh = blockIdx.x;
    const int b = bh / HEADS, h = bh % HEADS;
    const int w = b & 63;                       // window = b % nW (nW = 64)
    const int lane = threadIdx.x & 31;
    const int wave = threadIdx.x >> 5;          // query tile 0..3
    const int l16 = lane & 15, hi = lane >> 4;
    const size_t base = (size_t)bh * (NPAD * DH);

    Frag16 qf;
    {
        int row = wave * 16 + l16;
        int ko = hi * 16;
        const __bf16* qp = Qp + base + (size_t)row * DH;
        qf.h2[0] = *(const v8bf*)(qp + ko);
        qf.h2[1] = *(const v8bf*)(qp + ko + 8);
    }

    float s[4][8];
    #pragma unroll
    for (int t = 0; t < 4; ++t) {
        Frag16 kf;                               // B = K^T: lane = key column
        int col = t * 16 + l16;
        const __bf16* kp = Kp + base + (size_t)col * DH + hi * 16;
        kf.h2[0] = *(const v8bf*)(kp);
        kf.h2[1] = *(const v8bf*)(kp + 8);
        v8f a = {};
        a = __builtin_amdgcn_wmma_f32_16x16x32_bf16(false, qf.v, false, kf.v,
                                                    (short)0, a, false, false);
        bool colValid = col < SEQ_N;
        #pragma unroll
        for (int r = 0; r < 8; ++r) {
            int row = wave * 16 + (hi ? r + 8 : r);
            float v = a[r];
            if (colValid && row < SEQ_N)
                v += mask[((size_t)w * SEQ_N + row) * SEQ_N + col];
            if (!colValid) v = -1e30f;
            s[t][r] = v;
        }
    }

    // row softmax: each lane holds 8 rows x 1 col per tile; reduce across the
    // 16-lane half that shares those rows.
    #pragma unroll
    for (int r = 0; r < 8; ++r) {
        float m = fmaxf(fmaxf(s[0][r], s[1][r]), fmaxf(s[2][r], s[3][r]));
        #pragma unroll
        for (int off = 8; off >= 1; off >>= 1) m = fmaxf(m, __shfl_xor(m, off, 32));
        float e[4], sum = 0.f;
        #pragma unroll
        for (int t = 0; t < 4; ++t) { e[t] = __expf(s[t][r] - m); sum += e[t]; }
        #pragma unroll
        for (int off = 8; off >= 1; off >>= 1) sum += __shfl_xor(sum, off, 32);
        float inv = 1.0f / sum;
        int row = wave * 16 + (hi ? r + 8 : r);
        #pragma unroll
        for (int t = 0; t < 4; ++t)
            Ps[row * NPAD + t * 16 + l16] = (__bf16)(e[t] * inv);
    }
    // Same-wave LDS ops are in order (DScnt in-order per wave); waves touch
    // disjoint 16-row regions, so no block barrier is required here.

    v8f o0 = {}, o1 = {};
    #pragma unroll
    for (int kk = 0; kk < 2; ++kk) {            // keys 0..31, 32..63
        Frag16 pf, vf0, vf1;
        int row = wave * 16 + l16;
        int ko = kk * 32 + hi * 16;
        pf.h2[0] = *(const v8bf*)&Ps[row * NPAD + ko];
        pf.h2[1] = *(const v8bf*)&Ps[row * NPAD + ko + 8];
        const __bf16* vp = Vt + base + kk * 32 + hi * 16;
        vf0.h2[0] = *(const v8bf*)(vp + (size_t)l16 * NPAD);
        vf0.h2[1] = *(const v8bf*)(vp + (size_t)l16 * NPAD + 8);
        vf1.h2[0] = *(const v8bf*)(vp + (size_t)(l16 + 16) * NPAD);
        vf1.h2[1] = *(const v8bf*)(vp + (size_t)(l16 + 16) * NPAD + 8);
        o0 = __builtin_amdgcn_wmma_f32_16x16x32_bf16(false, pf.v, false, vf0.v,
                                                     (short)0, o0, false, false);
        o1 = __builtin_amdgcn_wmma_f32_16x16x32_bf16(false, pf.v, false, vf1.v,
                                                     (short)0, o1, false, false);
    }

    #pragma unroll
    for (int sub = 0; sub < 2; ++sub) {
        const v8f& o = sub ? o1 : o0;
        int c = h * DH + sub * 16 + l16;
        #pragma unroll
        for (int r = 0; r < 8; ++r) {
            int n = wave * 16 + (hi ? r + 8 : r);
            if (n < SEQ_N)
                Ao[((size_t)b * SEQ_N + n) * DIM_C + c] = (__bf16)o[r];
        }
    }
}

// ----------------------------------------------------------------- proj GEMM
__global__ void __launch_bounds__(256)
proj_gemm_kernel(const __bf16* __restrict__ Ao, const __bf16* __restrict__ wproj,
                 const float* __restrict__ proj_b, float* __restrict__ out) {
    __shared__ __bf16 As[64 * 32];
    __shared__ __bf16 Bs[64 * 32];
    const int mBase = blockIdx.x * 64;
    const int cBase = blockIdx.y * 64;
    const int tid   = threadIdx.x;
    const int wave  = tid >> 5, lane = tid & 31;
    const int wm = wave >> 1, wc = wave & 1;
    const int l16 = lane & 15, hi = lane >> 4;

    v8f acc0 = {}; v8f acc1 = {};

    for (int k0 = 0; k0 < DIM_C; k0 += 32) {
        {
            int idx = tid * 8;
            int r = idx >> 5, c = idx & 31;
            const __bf16* asrc = Ao + (size_t)(mBase + r) * DIM_C + k0 + c;
            *(v8bf*)&As[idx] = *(const v8bf*)asrc;
            const __bf16* bsrc = wproj + (size_t)(cBase + r) * DIM_C + k0 + c;
            *(v8bf*)&Bs[idx] = *(const v8bf*)bsrc;
            if (k0 + 32 < DIM_C) {
                __builtin_prefetch(asrc + 32, 0, 1);
                __builtin_prefetch(bsrc + 32, 0, 1);
            }
        }
        __syncthreads();
        Frag16 af, bf0, bf1;
        {
            int row = wm * 16 + l16;
            int ko = hi * 16;
            af.h2[0] = *(const v8bf*)&As[row * 32 + ko];
            af.h2[1] = *(const v8bf*)&As[row * 32 + ko + 8];
            int col0 = wc * 32 + l16;
            bf0.h2[0] = *(const v8bf*)&Bs[col0 * 32 + hi * 16];
            bf0.h2[1] = *(const v8bf*)&Bs[col0 * 32 + hi * 16 + 8];
            int col1 = col0 + 16;
            bf1.h2[0] = *(const v8bf*)&Bs[col1 * 32 + hi * 16];
            bf1.h2[1] = *(const v8bf*)&Bs[col1 * 32 + hi * 16 + 8];
        }
        acc0 = __builtin_amdgcn_wmma_f32_16x16x32_bf16(false, af.v, false, bf0.v,
                                                       (short)0, acc0, false, false);
        acc1 = __builtin_amdgcn_wmma_f32_16x16x32_bf16(false, af.v, false, bf1.v,
                                                       (short)0, acc1, false, false);
        __syncthreads();
    }

    #pragma unroll
    for (int sub = 0; sub < 2; ++sub) {
        const v8f& acc = sub ? acc1 : acc0;
        int j = cBase + wc * 32 + sub * 16 + l16;
        float bias = proj_b[j];
        #pragma unroll
        for (int r = 0; r < 8; ++r) {
            int t = mBase + wm * 16 + (hi ? r + 8 : r);
            out[(size_t)t * DIM_C + j] = acc[r] + bias;
        }
    }
}

// --------------------------------------------------------------------- launch
extern "C" void kernel_launch(void* const* d_in, const int* in_sizes, int n_in,
                              void* d_out, int out_size, void* d_ws, size_t ws_size,
                              hipStream_t stream) {
    const float* x      = (const float*)d_in[0];
    const float* mask   = (const float*)d_in[1];
    const float* qkv_w  = (const float*)d_in[2];
    const float* qkv_b  = (const float*)d_in[3];
    const float* proj_w = (const float*)d_in[4];
    const float* proj_b = (const float*)d_in[5];
    float* out = (float*)d_out;

    char* ws = (char*)d_ws;
    const size_t SZ_WQKV = (size_t)QKV_N * DIM_C * 2;        // 884736
    const size_t SZ_WPRJ = (size_t)DIM_C * DIM_C * 2;        // 294912
    const size_t SZ_QKV  = (size_t)BH * NPAD * DH * 2;       // 100663296 each
    __bf16* wqkv  = (__bf16*)ws;
    __bf16* wproj = (__bf16*)(ws + SZ_WQKV);
    __bf16* Qp    = (__bf16*)(ws + SZ_WQKV + SZ_WPRJ);
    __bf16* Kp    = (__bf16*)((char*)Qp + SZ_QKV);
    __bf16* Vt    = (__bf16*)((char*)Kp + SZ_QKV);
    __bf16* Ao    = (__bf16*)((char*)Vt + SZ_QKV);           // [tokens][384] bf16

    convert_w_kernel<<<(QKV_N * DIM_C + 255) / 256, 256, 0, stream>>>(qkv_w, proj_w, wqkv, wproj);
    pad_zero_kernel<<<BH, 256, 0, stream>>>(Qp, Kp, Vt);
    qkv_gemm_kernel<<<dim3(TOKENS / 64, QKV_N / 64), 256, 0, stream>>>(x, wqkv, qkv_b, Qp, Kp, Vt);
    attn_kernel<<<BH, 128, 0, stream>>>(Qp, Kp, Vt, mask, Ao);
    proj_gemm_kernel<<<dim3(TOKENS / 64, DIM_C / 64), 256, 0, stream>>>(Ao, wproj, proj_b, out);
}
